// V100MultiHeadAttention_944892805659
// MI455X (gfx1250) — matrix operand order repaired and checked by reading, not compiled, not run on hardware
//
#include <hip/hip_runtime.h>
#include <hip/hip_bf16.h>

#define BATCH 8
#define SEQ   2048
#define DIM   512
#define NH    8
#define HD    64
#define TTOT  (BATCH*SEQ)      // 16384 tokens
#define QKVC  (3*DIM)          // 1536

typedef __attribute__((ext_vector_type(16))) __bf16 v16bf;
typedef __attribute__((ext_vector_type(8)))  __bf16 v8bf;
typedef __attribute__((ext_vector_type(8)))  float  v8f;
typedef unsigned int v4u __attribute__((ext_vector_type(4)));
typedef int          v8i __attribute__((ext_vector_type(8)));
typedef int          v4i __attribute__((ext_vector_type(4)));

union BF16Frag { v16bf v; v8bf h[2]; };

__device__ __forceinline__ unsigned short f32_to_bf16(float f) {
  unsigned int u = __float_as_uint(f);
  u += 0x7fffu + ((u >> 16) & 1u);   // round to nearest even
  return (unsigned short)(u >> 16);
}

__device__ __forceinline__ v8f wmma_bf16(v16bf a, v16bf b, v8f c) {
  // D = A(16x32 bf16) * B(32x16 bf16) + C(16x16 f32)
  return __builtin_amdgcn_wmma_f32_16x16x32_bf16(false, a, false, b, (short)0, c,
                                                 false, false);
}

// ---- VALU cross-lane reductions over the 16-lane rows (wave32) ----
__device__ __forceinline__ float pl16(float x, unsigned s0, unsigned s1) {
  unsigned v = __float_as_uint(x);
  unsigned r = __builtin_amdgcn_permlane16(v, v, s0, s1, false, false);
  return __uint_as_float(r);
}
__device__ __forceinline__ float rowmax16(float x) {
  x = fmaxf(x, pl16(x, 0x67452301u, 0xEFCDAB89u)); // ^1
  x = fmaxf(x, pl16(x, 0x45670123u, 0xCDEF89ABu)); // ^2
  x = fmaxf(x, pl16(x, 0x32107654u, 0xBA98FEDCu)); // ^4
  x = fmaxf(x, pl16(x, 0xFEDCBA98u, 0x76543210u)); // ^8
  return x;
}
__device__ __forceinline__ float rowsum16(float x) {
  x += pl16(x, 0x67452301u, 0xEFCDAB89u);
  x += pl16(x, 0x45670123u, 0xCDEF89ABu);
  x += pl16(x, 0x32107654u, 0xBA98FEDCu);
  x += pl16(x, 0xFEDCBA98u, 0x76543210u);
  return x;
}

// A fragment 16x32 (M x K), row-major source with leading dim ld (elements).
__device__ __forceinline__ v16bf load_fragA(const unsigned short* base, int ld, int lane) {
  int r  = lane & 15;
  int kh = lane >> 4;
  const unsigned short* p = base + (size_t)r * ld + kh * 8;
  BF16Frag f;
  f.h[0] = *(const v8bf*)(p);
  f.h[1] = *(const v8bf*)(p + 16);
  return f.v;
}

// B fragment 32x16 (K x N): lane L holds column N=L%16, K in table order (8h+s / 16+8h+s-8).
__device__ __forceinline__ v16bf load_fragB(const unsigned short* base, int ld, int lane) {
  const unsigned short* p = base + (lane & 15);
  int kh = lane >> 4;
  BF16Frag f;
#pragma unroll
  for (int s = 0; s < 16; ++s) {
    int k = (s < 8) ? (kh * 8 + s) : (16 + kh * 8 + s - 8);
    f.v[s] = __builtin_bit_cast(__bf16, p[(size_t)k * ld]);
  }
  return f.v;
}

// ---------------- precision conversion kernels ----------------

__global__ void cvt_f32_bf16(const float* __restrict__ in,
                             unsigned short* __restrict__ out, int n) {
  int i = blockIdx.x * 256 + threadIdx.x;
  if (i < n) out[i] = f32_to_bf16(in[i]);
}

// in: R x C row-major f32  ->  out: C x R row-major bf16 (out[c][r] = in[r][c])
__global__ void cvt_transpose_f32_bf16(const float* __restrict__ in,
                                       unsigned short* __restrict__ out, int R, int C) {
  int i = blockIdx.x * 256 + threadIdx.x;
  if (i < R * C) {
    int orow = i / R;
    int ocol = i - orow * R;
    out[i] = f32_to_bf16(in[(size_t)ocol * C + orow]);
  }
}

// ---------------- WMMA GEMM: C[M,N] = A[M,K] * Wt[K,N] ----------------
template <bool OUT_F32>
__global__ void __launch_bounds__(128)
gemm_wmma(const unsigned short* __restrict__ A,
          const unsigned short* __restrict__ Wt,
          void* __restrict__ Cout, int N, int K) {
  int lane = threadIdx.x & 31;
  int wave = threadIdx.x >> 5;
  int mbase = blockIdx.x * 64 + (wave >> 1) * 32;
  int nbase = blockIdx.y * 128 + (wave & 1) * 64;

  v8f zero = {0.f,0.f,0.f,0.f,0.f,0.f,0.f,0.f};
  v8f acc[2][4];
#pragma unroll
  for (int i = 0; i < 2; ++i)
#pragma unroll
    for (int j = 0; j < 4; ++j) acc[i][j] = zero;

  for (int k0 = 0; k0 < K; k0 += 32) {
    v16bf a0 = load_fragA(A + (size_t)mbase * K + k0, K, lane);
    v16bf a1 = load_fragA(A + (size_t)(mbase + 16) * K + k0, K, lane);
#pragma unroll
    for (int j = 0; j < 4; ++j) {
      v16bf bj = load_fragB(Wt + (size_t)k0 * N + nbase + j * 16, N, lane);
      acc[0][j] = wmma_bf16(a0, bj, acc[0][j]);
      acc[1][j] = wmma_bf16(a1, bj, acc[1][j]);
    }
  }

  int cl = lane & 15;
  int rh = (lane >> 4) * 8;
#pragma unroll
  for (int mt = 0; mt < 2; ++mt)
#pragma unroll
    for (int j = 0; j < 4; ++j)
#pragma unroll
      for (int r = 0; r < 8; ++r) {
        size_t row = (size_t)(mbase + mt * 16 + r + rh);
        size_t idx = row * (size_t)N + nbase + j * 16 + cl;
        if (OUT_F32) ((float*)Cout)[idx] = acc[mt][j][r];
        else ((unsigned short*)Cout)[idx] = f32_to_bf16(acc[mt][j][r]);
      }
}

// ---------------- K head transpose: qkvb K-slice -> Kt[b,h,HD,SEQ] ----------------
__global__ void __launch_bounds__(256)
repack_kT(const unsigned short* __restrict__ qkvb, unsigned short* __restrict__ Kt) {
  __shared__ unsigned short tile[HD][HD + 8];
  int bh = blockIdx.x;
  int b = bh / NH, h = bh % NH;
  int s0 = blockIdx.y * 64;
  for (int i = threadIdx.x; i < 64 * 64; i += 256) {
    int sl = i >> 6, d = i & 63;
    tile[d][sl] = qkvb[(size_t)(b * SEQ + s0 + sl) * QKVC + DIM + h * HD + d];
  }
  __syncthreads();
  for (int i = threadIdx.x; i < 64 * 64; i += 256) {
    int d = i >> 6, sl = i & 63;
    Kt[((size_t)bh * HD + d) * SEQ + s0 + sl] = tile[d][sl];
  }
}

// ---------------- TDM 2-D tile load: global -> LDS (TENSORcnt) ----------------
// D# per CDNA5 ISA 8.3/8.4: group0 {count=1, lds_addr, global_addr[56:0], type=2},
// group1 {data_size=2B, tensor_dim0/1, tile_dim0/1, tensor_dim0_stride}.
// This toolchain exposes the 6-arg builtin: (v4u, v8i, v4i, v4i, v8i, i32 cpol).
__device__ __forceinline__ void tdm_load_2d(const void* gaddr, unsigned lds_off,
                                            unsigned tensor_d0, unsigned tensor_d1,
                                            unsigned tile_d0, unsigned tile_d1,
                                            unsigned d0_stride) {
  unsigned long long ga = (unsigned long long)(size_t)gaddr;
  v4u g0;
  g0[0] = 1u;                                        // count=1, user mode
  g0[1] = lds_off;                                   // lds_addr (bytes)
  g0[2] = (unsigned)(ga & 0xffffffffu);              // global_addr[31:0]
  g0[3] = (unsigned)((ga >> 32) & 0x01ffffffu)       // global_addr[56:32]
        | (2u << 30);                                // type = 2 ("image")
  v8i g1;
  g1[0] = (int)(1u << 16);                           // data_size=1 -> 2 bytes
  g1[1] = (int)((tensor_d0 & 0xffffu) << 16);        // tensor_dim0[15:0] @ 63:48
  g1[2] = (int)(((tensor_d0 >> 16) & 0xffffu) |      // tensor_dim0[31:16]
                ((tensor_d1 & 0xffffu) << 16));      // tensor_dim1[15:0]
  g1[3] = (int)(((tensor_d1 >> 16) & 0xffffu) |      // tensor_dim1[31:16]
                ((tile_d0 & 0xffffu) << 16));        // tile_dim0 @ 127:112
  g1[4] = (int)(tile_d1 & 0xffffu);                  // tile_dim1; tile_dim2=0
  g1[5] = (int)d0_stride;                            // tensor_dim0_stride[31:0]
  g1[6] = 0;                                         // stride hi / dim1_stride lo
  g1[7] = 0;
  v4i gz4 = {0, 0, 0, 0};
  v8i gz8 = {0, 0, 0, 0, 0, 0, 0, 0};
  __builtin_amdgcn_tensor_load_to_lds(g0, g1, gz4, gz4, gz8, 0);
}

// ---------------- flash attention: TDM K tile + async V tile ----------------
// Block = 256 threads = 8 waves; each wave owns 16 query rows (128 rows/block).
// Per 32-key tile: K^T tile (64x32) moved by the Tensor Data Mover, V tile
// (32x64) by async global->LDS b128 copies; double-buffered, shared by 8 waves
// (8x less L2 traffic than per-wave streaming). Wave 0 drives the DMA and
// waits TENSORcnt/ASYNCcnt; the workgroup barrier publishes the tiles.
__device__ __forceinline__ void issue_tiles(const unsigned short* ktp,
                                            const unsigned short* vbase,
                                            unsigned short* kdst,
                                            unsigned short* vdst,
                                            int kt, int lane) {
  // K^T tile via TDM: tensor [HD x SEQ], tile 32(keys) x 64(rows)
  tdm_load_2d(ktp + kt, (unsigned)(size_t)kdst,
              SEQ, HD, /*tile_d0=*/32, /*tile_d1=*/64, /*stride=*/SEQ);
  // V tile via async copies: 32 rows (key) x 64 d, 128B/row = 8 x 16B chunks
  unsigned long long vb = (unsigned long long)(size_t)vbase;
#pragma unroll
  for (int it = 0; it < 8; ++it) {
    int c = it * 32 + lane;
    int vrow = c >> 3, vcol = (c & 7) * 8;
    unsigned vld = (unsigned)(size_t)(vdst + vrow * 64 + vcol);
    unsigned vgo = (unsigned)(((size_t)(kt + vrow) * QKVC + vcol) * 2);
    asm volatile("global_load_async_to_lds_b128 %0, %1, %2"
                 :: "v"(vld), "v"(vgo), "s"(vb) : "memory");
  }
}

__global__ void __launch_bounds__(256)
flash_attn(const unsigned short* __restrict__ qkvb,
           const unsigned short* __restrict__ Kt,
           unsigned short* __restrict__ attnb) {
  __shared__ __attribute__((aligned(16))) unsigned short kbuf[2][64][32]; // K^T tile
  __shared__ __attribute__((aligned(16))) unsigned short vbuf[2][32][64]; // V tile
  __shared__ __attribute__((aligned(16))) unsigned short plds[8][16][40]; // P restage

  int lane = threadIdx.x & 31;
  int wave = threadIdx.x >> 5;
  int bh = blockIdx.x;
  int b = bh >> 3, h = bh & 7;
  int q0 = blockIdx.y * 128 + wave * 16;

  const unsigned short* qbase = qkvb + (size_t)(b * SEQ + q0) * QKVC + h * HD;
  const unsigned short* ktp   = Kt + (size_t)bh * HD * SEQ;
  const unsigned short* vbase = qkvb + (size_t)(b * SEQ) * QKVC + 2 * DIM + h * HD;

  v16bf qa[2];
  qa[0] = load_fragA(qbase, QKVC, lane);
  qa[1] = load_fragA(qbase + 32, QKVC, lane);

  v8f zero = {0.f,0.f,0.f,0.f,0.f,0.f,0.f,0.f};
  v8f o[4];
#pragma unroll
  for (int j = 0; j < 4; ++j) o[j] = zero;
  float rmax[8], rsum[8];
#pragma unroll
  for (int r = 0; r < 8; ++r) { rmax[r] = -3.0e38f; rsum[r] = 0.f; }

  // softmax in exp2 domain: scale = HD^-0.5 * log2(e)
  const float SC = 0.125f * 1.4426950408889634f;

  int cl = lane & 15;
  int kh = lane >> 4;
  unsigned short (*pl)[40] = plds[wave];

  int cur = 0;
  if (wave == 0)
    issue_tiles(ktp, vbase, &kbuf[0][0][0], &vbuf[0][0][0], 0, lane);

  for (int kt = 0; kt < SEQ; kt += 32) {
    if (wave == 0) {
      __builtin_amdgcn_s_wait_tensorcnt(0);
      asm volatile("s_wait_asynccnt 0x0" ::: "memory");
    }
    __syncthreads();                     // tile `cur` ready; prev compute done
    int nxt = cur ^ 1;
    if (wave == 0 && kt + 32 < SEQ)
      issue_tiles(ktp, vbase, &kbuf[nxt][0][0], &vbuf[nxt][0][0], kt + 32, lane);

    // ---- scores: S = Q(16x64) @ K^T(64x32), from LDS K^T tile ----
    v8f s0 = zero, s1 = zero;
#pragma unroll
    for (int kk = 0; kk < 2; ++kk) {
      BF16Frag kb0, kb1;                 // B col = key = cl, K = d in table order
#pragma unroll
      for (int s = 0; s < 16; ++s) {
        int d = kk * 32 + ((s < 8) ? (kh * 8 + s) : (16 + kh * 8 + s - 8));
        kb0.v[s] = __builtin_bit_cast(__bf16, kbuf[cur][d][cl]);
        kb1.v[s] = __builtin_bit_cast(__bf16, kbuf[cur][d][16 + cl]);
      }
      s0 = wmma_bf16(qa[kk], kb0.v, s0);
      s1 = wmma_bf16(qa[kk], kb1.v, s1);
    }

    // ---- online softmax; reductions via v_permlane16 (pure VALU) ----
#pragma unroll
    for (int r = 0; r < 8; ++r) {
      float a = s0[r] * SC;
      float c = s1[r] * SC;
      float mx = rowmax16(fmaxf(a, c));
      float nm = fmaxf(rmax[r], mx);
      float fac = __builtin_amdgcn_exp2f(rmax[r] - nm);
      rmax[r] = nm;
      rsum[r] *= fac;
      o[0][r] *= fac; o[1][r] *= fac; o[2][r] *= fac; o[3][r] *= fac;
      float p0 = __builtin_amdgcn_exp2f(a - nm);
      float p1 = __builtin_amdgcn_exp2f(c - nm);
      rsum[r] += rowsum16(p0 + p1);
      int m = r + (kh << 3);             // C layout: M = r + 8*(lane/16)
      pl[m][cl] = f32_to_bf16(p0);
      pl[m][16 + cl] = f32_to_bf16(p1);
    }
    asm volatile("s_wait_dscnt 0" ::: "memory");  // wave-internal LDS RAW
    BF16Frag pf;                         // re-read P as A-layout (M = cl)
    pf.h[0] = *(const v8bf*)&pl[cl][kh * 8];
    pf.h[1] = *(const v8bf*)&pl[cl][16 + kh * 8];

    // ---- O += P(16x32) @ V(32x64), V tile from LDS ----
#pragma unroll
    for (int j = 0; j < 4; ++j) {
      BF16Frag vf;                       // B col = d = j*16+cl, K = key in table order
#pragma unroll
      for (int s = 0; s < 16; ++s) {
        int key = (s < 8) ? (kh * 8 + s) : (16 + kh * 8 + s - 8);
        vf.v[s] = __builtin_bit_cast(__bf16, vbuf[cur][key][j * 16 + cl]);
      }
      o[j] = wmma_bf16(pf.v, vf.v, o[j]);
    }
    cur = nxt;
  }

#pragma unroll
  for (int r = 0; r < 8; ++r) {
    float inv = 1.0f / rsum[r];
    int m = r + (kh << 3);
    size_t row = (size_t)(b * SEQ + q0 + m) * DIM + h * HD;
#pragma unroll
    for (int j = 0; j < 4; ++j)
      attnb[row + j * 16 + cl] = f32_to_bf16(o[j][r] * inv);
  }
}

// ---------------- launcher ----------------

extern "C" void kernel_launch(void* const* d_in, const int* in_sizes, int n_in,
                              void* d_out, int out_size, void* d_ws, size_t ws_size,
                              hipStream_t stream) {
  const float* x    = (const float*)d_in[0];
  // d_in[1] = cu_seqlens, d_in[2] = max_seqlen : uniform S, unused
  const float* Wqkv = (const float*)d_in[3];
  const float* Wout = (const float*)d_in[4];
  float* out = (float*)d_out;

  char* ws = (char*)d_ws;
  size_t off = 0;
  auto take = [&](size_t bytes) -> unsigned short* {
    unsigned short* p = (unsigned short*)(ws + off);
    off += (bytes + 255) & ~(size_t)255;
    return p;
  };
  unsigned short* xb     = take((size_t)TTOT * DIM * 2);            // 16 MB
  unsigned short* wqkvT  = take((size_t)QKVC * DIM * 2);            // 1.5 MB [512,1536]
  unsigned short* woutT  = take((size_t)DIM * DIM * 2);             // 0.5 MB [512,512]
  unsigned short* qkvb   = take((size_t)TTOT * QKVC * 2);           // 50 MB
  unsigned short* Kt     = take((size_t)BATCH * NH * HD * SEQ * 2); // 16 MB
  unsigned short* attnb  = take((size_t)TTOT * DIM * 2);            // 16 MB

  int nx = TTOT * DIM;
  cvt_f32_bf16<<<(nx + 255) / 256, 256, 0, stream>>>(x, xb, nx);
  cvt_transpose_f32_bf16<<<(QKVC * DIM + 255) / 256, 256, 0, stream>>>(Wqkv, wqkvT, QKVC, DIM);
  cvt_transpose_f32_bf16<<<(DIM * DIM + 255) / 256, 256, 0, stream>>>(Wout, woutT, DIM, DIM);

  // qkv = x @ Wqkv^T  (M=16384, N=1536, K=512)
  gemm_wmma<false><<<dim3(TTOT / 64, QKVC / 128), 128, 0, stream>>>(xb, wqkvT, qkvb, QKVC, DIM);

  // K head transpose for QK^T B-fragments
  repack_kT<<<dim3(BATCH * NH, SEQ / 64), 256, 0, stream>>>(qkvb, Kt);

  // streaming-softmax attention, TDM + async double-buffered K/V tiles
  flash_attn<<<dim3(BATCH * NH, SEQ / 128), 256, 0, stream>>>(qkvb, Kt, attnb);

  // out = attn @ Wout^T  (M=16384, N=512, K=512), f32 output
  gemm_wmma<true><<<dim3(TTOT / 64, DIM / 128), 128, 0, stream>>>(attnb, woutT, out, DIM, DIM);
}
